// MultiAttention_10531259810577
// MI455X (gfx1250) — matrix operand order repaired and checked by reading, not compiled, not run on hardware
//
#include <hip/hip_runtime.h>

// Problem constants (from reference): B=8, Lp=Lq=2048, DIN=H=DREP=1024
#define BB    8
#define LPN   2048
#define LQN   2048
#define DINN  1024
#define HHN   1024
#define DREPN 1024

// Tiling
#define TM 128
#define TN 128
#define KT 64
#define LS 72   // padded LDS row stride in bf16 elements (144B -> conflict-free)

typedef __attribute__((ext_vector_type(16))) __bf16 v16bf;
typedef __attribute__((ext_vector_type(8)))  __bf16 v8bf;
typedef __attribute__((ext_vector_type(8)))  float  v8f;

#define NEG_INF (-__builtin_inff())

// ---- CDNA5 async global->LDS path (guarded: falls back to sync copies) ----
#if defined(__has_builtin)
#if __has_builtin(__builtin_amdgcn_global_load_async_to_lds_b128) && \
    __has_builtin(__builtin_amdgcn_s_wait_asynccnt)
#define HAVE_ASYNC_LDS 1
#endif
#endif
#ifndef HAVE_ASYNC_LDS
#define HAVE_ASYNC_LDS 0
#endif

#if HAVE_ASYNC_LDS
// Probe told us: param 0 is a pointer to 'int __attribute__((vector_size(16)))'
// (AS1), so route the cast through typedefs where the address space binds to
// the pointee type.
typedef __attribute__((ext_vector_type(4))) int i32x4;
typedef __attribute__((address_space(1))) i32x4* as1_i32x4p;
typedef __attribute__((address_space(3))) i32x4* as3_i32x4p;

__device__ __forceinline__ void async_cp16(const __bf16* g, __bf16* l) {
    i32x4* gg = (i32x4*)g;   // const-cast + reinterpret (generic AS)
    i32x4* ll = (i32x4*)l;
    __builtin_amdgcn_global_load_async_to_lds_b128(
        (as1_i32x4p)gg, (as3_i32x4p)ll, 0, 0);
}
#endif

__device__ __forceinline__ v8f vzero8() {
    v8f z;
#pragma unroll
    for (int i = 0; i < 8; ++i) z[i] = 0.0f;
    return z;
}

// Load a 16-element bf16 fragment as two 16B LDS reads (ds_load_b128 x2).
__device__ __forceinline__ v16bf frag_ld(const __bf16* base, int second_off) {
    v8bf lo = *(const v8bf*)base;
    v8bf hi = *(const v8bf*)(base + second_off);
    v16bf r;
#pragma unroll
    for (int i = 0; i < 8; ++i) { r[i] = lo[i]; r[8 + i] = hi[i]; }
    return r;
}

// A fragment (16x32, MxK): lane half h reads K = k0 + h*8 + [0..7] and +16.
__device__ __forceinline__ v16bf frag_A(const __bf16* s, int row, int k0, int half) {
    return frag_ld(s + row * LS + k0 + half * 8, 16);
}
// B fragment (32x16, KxN, stored N-major): lane half h reads K = k0 + h*8 + [0..7] and +16.
__device__ __forceinline__ v16bf frag_B(const __bf16* s, int nrow, int k0, int half) {
    return frag_ld(s + nrow * LS + k0 + half * 8, 16);
}

// Online (max, sum-of-exp) merge; treats (om, os) as a partial softmax state.
__device__ __forceinline__ void smerge(float& m, float& s, float om, float os) {
    float nm = fmaxf(m, om);
    float a = (s  > 0.0f) ? s  * __expf(m  - nm) : 0.0f;
    float b = (os > 0.0f) ? os * __expf(om - nm) : 0.0f;
    s = a + b;
    m = nm;
}

// ---------------------------------------------------------------------------
// Kernel 1: projection GEMM. Rows [0, B*Lp) from p, [B*Lp, B*(Lp+Lq)) from q.
// out = relu(x @ Uw^T) (* D for q rows), stored bf16.
// Register-staged double-buffered LDS: global loads for tile t+1 overlap the
// WMMAs of tile t; one barrier per K-iteration.
// grid = (B*(Lp+Lq)/128, H/128), block = 256 (8 waves, 4x2 wave grid)
// ---------------------------------------------------------------------------
__global__ __launch_bounds__(256) void proj_kernel(
    const float* __restrict__ p, const float* __restrict__ q,
    const float* __restrict__ Uw, const float* __restrict__ D,
    __bf16* __restrict__ Up, __bf16* __restrict__ Uq)
{
    __shared__ __bf16 As[2][TM * LS];
    __shared__ __bf16 Bs[2][TN * LS];

    const int tid  = threadIdx.x;
    const int row0 = blockIdx.x * TM;     // concat row space
    const int col0 = blockIdx.y * TN;     // over H
    const bool is_q = row0 >= BB * LPN;
    const int  l    = is_q ? row0 - BB * LPN : row0;   // b*2048 + m base
    const float* __restrict__ src = is_q ? q : p;
    __bf16* __restrict__ dst      = is_q ? Uq : Up;

    const int wave = tid >> 5, lane = tid & 31;
    const int wm = wave & 3, wn = wave >> 2;
    const int half = lane >> 4, lr = lane & 15;

    v8f acc[2][4];
#pragma unroll
    for (int i = 0; i < 2; ++i)
#pragma unroll
        for (int j = 0; j < 4; ++j) acc[i][j] = vzero8();

    float4 ra[8], rb[8];
    auto load_regs = [&](int kt) {
#pragma unroll
        for (int it = 0; it < 8; ++it) {
            int idx = tid + it * 256;
            int r = idx >> 4, cv = (idx & 15) * 4;
            ra[it] = *(const float4*)(src + (size_t)(l + r) * DINN + kt + cv);
            rb[it] = *(const float4*)(Uw + (size_t)(col0 + r) * DINN + kt + cv);
        }
    };
    auto store_lds = [&](int buf) {
#pragma unroll
        for (int it = 0; it < 8; ++it) {
            int idx = tid + it * 256;
            int r = idx >> 4, cv = (idx & 15) * 4;
            __bf16* da = &As[buf][r * LS + cv];
            da[0] = (__bf16)ra[it].x; da[1] = (__bf16)ra[it].y;
            da[2] = (__bf16)ra[it].z; da[3] = (__bf16)ra[it].w;
            __bf16* db = &Bs[buf][r * LS + cv];
            db[0] = (__bf16)rb[it].x; db[1] = (__bf16)rb[it].y;
            db[2] = (__bf16)rb[it].z; db[3] = (__bf16)rb[it].w;
        }
    };

    load_regs(0);
    store_lds(0);
    __syncthreads();

    for (int kt = 0, t = 0; kt < DINN; kt += KT, ++t) {
        const int cur = t & 1;
        const bool has_next = (kt + KT) < DINN;
        if (has_next) load_regs(kt + KT);   // overlap with WMMAs below
#pragma unroll
        for (int k0 = 0; k0 < KT; k0 += 32) {
            v16bf a0 = frag_A(As[cur], wm * 32 + lr,      k0, half);
            v16bf a1 = frag_A(As[cur], wm * 32 + 16 + lr, k0, half);
            v16bf bf[4];
#pragma unroll
            for (int j = 0; j < 4; ++j) bf[j] = frag_B(Bs[cur], wn * 64 + j * 16 + lr, k0, half);
#pragma unroll
            for (int j = 0; j < 4; ++j) {
                acc[0][j] = __builtin_amdgcn_wmma_f32_16x16x32_bf16(false, a0, false, bf[j], (short)0, acc[0][j], false, false);
                acc[1][j] = __builtin_amdgcn_wmma_f32_16x16x32_bf16(false, a1, false, bf[j], (short)0, acc[1][j], false, false);
            }
        }
        if (has_next) store_lds(cur ^ 1);
        __syncthreads();
    }
    // Epilogue: relu (+ D scale for q rows), store bf16
#pragma unroll
    for (int i = 0; i < 2; ++i)
#pragma unroll
        for (int j = 0; j < 4; ++j) {
            int n = col0 + wn * 64 + j * 16 + lr;
            float dsc = is_q ? D[n] : 1.0f;
#pragma unroll
            for (int e = 0; e < 8; ++e) {
                int mr = wm * 32 + i * 16 + half * 8 + e;
                float v = fmaxf(acc[i][j][e], 0.0f) * dsc;
                dst[(size_t)(l + mr) * HHN + n] = (__bf16)v;
            }
        }
}

// ---------------------------------------------------------------------------
// Kernel 2: scores S[b] = Up[b] @ Uq[b]^T, bf16 in, bf16 out.
// Pure bf16 copy tiles -> use CDNA5 async global->LDS (ASYNCcnt) if the
// toolchain exposes the builtin; else register-staged double buffering.
// grid = (16, 16, B)
// ---------------------------------------------------------------------------
__global__ __launch_bounds__(256) void scores_kernel(
    const __bf16* __restrict__ Up, const __bf16* __restrict__ Uq,
    __bf16* __restrict__ S)
{
    __shared__ __bf16 As[2][TM * LS];
    __shared__ __bf16 Bs[2][TN * LS];

    const int tid = threadIdx.x;
    const int b   = blockIdx.z;
    const int m0  = blockIdx.x * TM;
    const int n0  = blockIdx.y * TN;
    const __bf16* __restrict__ Ag = Up + (size_t)b * LPN * HHN;
    const __bf16* __restrict__ Bg = Uq + (size_t)b * LQN * HHN;

    const int wave = tid >> 5, lane = tid & 31;
    const int wm = wave & 3, wn = wave >> 2;
    const int half = lane >> 4, lr = lane & 15;

    v8f acc[2][4];
#pragma unroll
    for (int i = 0; i < 2; ++i)
#pragma unroll
        for (int j = 0; j < 4; ++j) acc[i][j] = vzero8();

#if HAVE_ASYNC_LDS
    auto issue_tile = [&](int kt, int buf) {
#pragma unroll
        for (int it = 0; it < 4; ++it) {
            int idx = tid + it * 256;
            int r = idx >> 3, c8 = (idx & 7) * 8;
            async_cp16(Ag + (size_t)(m0 + r) * HHN + kt + c8, &As[buf][r * LS + c8]);
            async_cp16(Bg + (size_t)(n0 + r) * HHN + kt + c8, &Bs[buf][r * LS + c8]);
        }
    };
    issue_tile(0, 0);
    __builtin_amdgcn_s_wait_asynccnt(0);
    __syncthreads();
#else
    v8bf sa[4], sb[4];
    auto load_regs = [&](int kt) {
#pragma unroll
        for (int it = 0; it < 4; ++it) {
            int idx = tid + it * 256;
            int r = idx >> 3, c8 = (idx & 7) * 8;
            sa[it] = *(const v8bf*)(Ag + (size_t)(m0 + r) * HHN + kt + c8);
            sb[it] = *(const v8bf*)(Bg + (size_t)(n0 + r) * HHN + kt + c8);
        }
    };
    auto store_lds = [&](int buf) {
#pragma unroll
        for (int it = 0; it < 4; ++it) {
            int idx = tid + it * 256;
            int r = idx >> 3, c8 = (idx & 7) * 8;
            *(v8bf*)(&As[buf][r * LS + c8]) = sa[it];
            *(v8bf*)(&Bs[buf][r * LS + c8]) = sb[it];
        }
    };
    load_regs(0);
    store_lds(0);
    __syncthreads();
#endif

    for (int kt = 0, t = 0; kt < HHN; kt += KT, ++t) {
        const int cur = t & 1;
        const bool has_next = (kt + KT) < HHN;
#if HAVE_ASYNC_LDS
        if (has_next) issue_tile(kt + KT, cur ^ 1);   // DMA next tile during WMMAs
#else
        if (has_next) load_regs(kt + KT);
#endif
#pragma unroll
        for (int k0 = 0; k0 < KT; k0 += 32) {
            v16bf a0 = frag_A(As[cur], wm * 32 + lr,      k0, half);
            v16bf a1 = frag_A(As[cur], wm * 32 + 16 + lr, k0, half);
            v16bf bf[4];
#pragma unroll
            for (int j = 0; j < 4; ++j) bf[j] = frag_B(Bs[cur], wn * 64 + j * 16 + lr, k0, half);
#pragma unroll
            for (int j = 0; j < 4; ++j) {
                acc[0][j] = __builtin_amdgcn_wmma_f32_16x16x32_bf16(false, a0, false, bf[j], (short)0, acc[0][j], false, false);
                acc[1][j] = __builtin_amdgcn_wmma_f32_16x16x32_bf16(false, a1, false, bf[j], (short)0, acc[1][j], false, false);
            }
        }
#if HAVE_ASYNC_LDS
        if (has_next) __builtin_amdgcn_s_wait_asynccnt(0);
#else
        if (has_next) store_lds(cur ^ 1);
#endif
        __syncthreads();
    }

    __bf16* __restrict__ Sb = S + (size_t)b * LPN * LQN;
#pragma unroll
    for (int i = 0; i < 2; ++i)
#pragma unroll
        for (int j = 0; j < 4; ++j) {
            int n = n0 + wn * 64 + j * 16 + lr;
#pragma unroll
            for (int e = 0; e < 8; ++e) {
                int mr = wm * 32 + i * 16 + half * 8 + e;
                Sb[(size_t)(m0 + mr) * LQN + n] = (__bf16)acc[i][j][e];
            }
        }
}

// ---------------------------------------------------------------------------
// Kernel 3: row-direction softmax stats (softmax over Lq, mask=q_mask).
// One wave per row; wave32 shfl_xor reduction. grid = B*Lp/8 = 2048.
// ---------------------------------------------------------------------------
__global__ __launch_bounds__(256) void stats_row_kernel(
    const __bf16* __restrict__ S, const unsigned char* __restrict__ qmask,
    float* __restrict__ stats)
{
    const int wave = threadIdx.x >> 5, lane = threadIdx.x & 31;
    const int row = blockIdx.x * 8 + wave;        // b*2048 + m
    const int b = row >> 11;
    const __bf16* __restrict__ Sr = S + (size_t)row * LQN;
    const unsigned char* __restrict__ mk = qmask + (size_t)b * LQN;

    float mx = NEG_INF, sm = 0.0f;
    for (int j = lane; j < LQN; j += 32)
        if (!mk[j]) smerge(mx, sm, (float)Sr[j], 1.0f);
#pragma unroll
    for (int off = 16; off > 0; off >>= 1) {
        float om = __shfl_xor(mx, off, 32);
        float os = __shfl_xor(sm, off, 32);
        smerge(mx, sm, om, os);
    }
    if (lane == 0) { stats[(size_t)row * 2] = mx; stats[(size_t)row * 2 + 1] = sm; }
}

// ---------------------------------------------------------------------------
// Kernel 4a/4b: column-direction stats (softmax over Lp, mask=p_mask),
// split over m-chunks for parallelism, then combined.
// ---------------------------------------------------------------------------
__global__ __launch_bounds__(256) void stats_col_partial_kernel(
    const __bf16* __restrict__ S, const unsigned char* __restrict__ pmask,
    float* __restrict__ partial)
{
    const int n = blockIdx.x * 256 + threadIdx.x;  // column
    const int mc = blockIdx.y;                     // m-chunk
    const int b = blockIdx.z;
    const __bf16* __restrict__ Sb = S + (size_t)b * LPN * LQN;
    const unsigned char* __restrict__ mk = pmask + (size_t)b * LPN;

    float mx = NEG_INF, sm = 0.0f;
    const int m0 = mc * 256;
    for (int m = m0; m < m0 + 256; ++m)
        if (!mk[m]) smerge(mx, sm, (float)Sb[(size_t)m * LQN + n], 1.0f);
    size_t o = (((size_t)mc * BB + b) * LQN + n) * 2;
    partial[o] = mx; partial[o + 1] = sm;
}

__global__ __launch_bounds__(256) void stats_col_combine_kernel(
    const float* __restrict__ partial, float* __restrict__ stats)
{
    const int gid = blockIdx.x * 256 + threadIdx.x;  // b*2048 + n
    const int b = gid >> 11, n = gid & 2047;
    float mx = NEG_INF, sm = 0.0f;
#pragma unroll
    for (int c = 0; c < 8; ++c) {
        size_t o = (((size_t)c * BB + b) * LQN + n) * 2;
        smerge(mx, sm, partial[o], partial[o + 1]);
    }
    stats[(size_t)gid * 2] = mx; stats[(size_t)gid * 2 + 1] = sm;
}

// ---------------------------------------------------------------------------
// Kernel 5: attention apply. out[b,i,d] = sum_j w(i,j) * V[b,j,d],
// w = mask[j] ? 0 : exp(S(i,j) - max_i) / sum_i, S read transposed for q->p.
// Register-staged double-buffered LDS (transforms prevent async copy).
// grid = (2048/128, DREP/128, B)
// ---------------------------------------------------------------------------
__global__ __launch_bounds__(256) void attn_kernel(
    const __bf16* __restrict__ S, const float* __restrict__ stats,
    const unsigned char* __restrict__ mask, const float* __restrict__ V,
    float* __restrict__ out, int trans)
{
    __shared__ __bf16 As[2][TM * LS];   // weights: [i][j]
    __shared__ __bf16 Bs[2][TN * LS];   // values transposed: [d][j]

    const int tid = threadIdx.x;
    const int b  = blockIdx.z;
    const int i0 = blockIdx.x * TM;
    const int d0 = blockIdx.y * TN;
    const __bf16* __restrict__ Sb = S + (size_t)b * LPN * LQN;
    const float* __restrict__ statsb = stats + (size_t)b * 2048 * 2;
    const unsigned char* __restrict__ maskb = mask + (size_t)b * 2048;
    const float* __restrict__ Vb = V + (size_t)b * 2048 * DREPN;

    const int wave = tid >> 5, lane = tid & 31;
    const int wm = wave & 3, wn = wave >> 2;
    const int half = lane >> 4, lr = lane & 15;

    v8f acc[2][4];
#pragma unroll
    for (int i = 0; i < 2; ++i)
#pragma unroll
        for (int j = 0; j < 4; ++j) acc[i][j] = vzero8();

    // Direction-1 softmax stats are j0-invariant per thread: hoist them.
    float mxi[4], isv[4];
    if (!trans) {
#pragma unroll
        for (int it = 0; it < 4; ++it) {
            int gi = i0 + ((tid + it * 256) >> 3);
            mxi[it] = statsb[(size_t)gi * 2];
            isv[it] = 1.0f / statsb[(size_t)gi * 2 + 1];
        }
    }

    v8bf rs[4];
    float4 rv[8];
    auto load_regs = [&](int j0) {
        if (!trans) {
#pragma unroll
            for (int it = 0; it < 4; ++it) {
                int idx = tid + it * 256;
                int r = idx >> 3, c8 = (idx & 7) * 8;
                rs[it] = *(const v8bf*)(Sb + (size_t)(i0 + r) * LQN + j0 + c8);
            }
        } else {
#pragma unroll
            for (int it = 0; it < 4; ++it) {
                int idx = tid + it * 256;
                int jr = idx >> 4, c8 = (idx & 15) * 8;
                rs[it] = *(const v8bf*)(Sb + (size_t)(j0 + jr) * LQN + i0 + c8);
            }
        }
#pragma unroll
        for (int it = 0; it < 8; ++it) {
            int idx = tid + it * 256;
            int jr = idx >> 5, c4 = (idx & 31) * 4;
            rv[it] = *(const float4*)(Vb + (size_t)(j0 + jr) * DREPN + d0 + c4);
        }
    };
    auto store_lds = [&](int j0, int buf) {
        if (!trans) {
#pragma unroll
            for (int it = 0; it < 4; ++it) {
                int idx = tid + it * 256;
                int r = idx >> 3, c8 = (idx & 7) * 8;
                __bf16* d = &As[buf][r * LS + c8];
#pragma unroll
                for (int e = 0; e < 8; ++e) {
                    float w = maskb[j0 + c8 + e] ? 0.0f
                              : __expf((float)rs[it][e] - mxi[it]) * isv[it];
                    d[e] = (__bf16)w;
                }
            }
        } else {
#pragma unroll
            for (int it = 0; it < 4; ++it) {
                int idx = tid + it * 256;
                int jr = idx >> 4, c8 = (idx & 15) * 8;
                bool mk = maskb[j0 + jr];
#pragma unroll
                for (int e = 0; e < 8; ++e) {
                    int gi = i0 + c8 + e;
                    float w = mk ? 0.0f
                              : __expf((float)rs[it][e] - statsb[(size_t)gi * 2])
                                    * (1.0f / statsb[(size_t)gi * 2 + 1]);
                    As[buf][(c8 + e) * LS + jr] = (__bf16)w;
                }
            }
        }
#pragma unroll
        for (int it = 0; it < 8; ++it) {
            int idx = tid + it * 256;
            int jr = idx >> 5, c4 = (idx & 31) * 4;
            Bs[buf][(c4 + 0) * LS + jr] = (__bf16)rv[it].x;
            Bs[buf][(c4 + 1) * LS + jr] = (__bf16)rv[it].y;
            Bs[buf][(c4 + 2) * LS + jr] = (__bf16)rv[it].z;
            Bs[buf][(c4 + 3) * LS + jr] = (__bf16)rv[it].w;
        }
    };

    load_regs(0);
    store_lds(0, 0);
    __syncthreads();

    for (int j0 = 0, t = 0; j0 < 2048; j0 += KT, ++t) {
        const int cur = t & 1;
        const bool has_next = (j0 + KT) < 2048;
        if (has_next) load_regs(j0 + KT);   // overlap with WMMAs below
#pragma unroll
        for (int k0 = 0; k0 < KT; k0 += 32) {
            v16bf a0 = frag_A(As[cur], wm * 32 + lr,      k0, half);
            v16bf a1 = frag_A(As[cur], wm * 32 + 16 + lr, k0, half);
            v16bf bf[4];
#pragma unroll
            for (int j = 0; j < 4; ++j) bf[j] = frag_B(Bs[cur], wn * 64 + j * 16 + lr, k0, half);
#pragma unroll
            for (int j = 0; j < 4; ++j) {
                acc[0][j] = __builtin_amdgcn_wmma_f32_16x16x32_bf16(false, a0, false, bf[j], (short)0, acc[0][j], false, false);
                acc[1][j] = __builtin_amdgcn_wmma_f32_16x16x32_bf16(false, a1, false, bf[j], (short)0, acc[1][j], false, false);
            }
        }
        if (has_next) store_lds(j0 + KT, cur ^ 1);
        __syncthreads();
    }
#pragma unroll
    for (int i = 0; i < 2; ++i)
#pragma unroll
        for (int j = 0; j < 4; ++j) {
            int n = d0 + wn * 64 + j * 16 + lr;
#pragma unroll
            for (int e = 0; e < 8; ++e) {
                int mr = wm * 32 + i * 16 + half * 8 + e;
                out[(size_t)b * 2048 * DREPN + (size_t)(i0 + mr) * DREPN + n] = acc[i][j][e];
            }
        }
}

// ---------------------------------------------------------------------------
extern "C" void kernel_launch(void* const* d_in, const int* in_sizes, int n_in,
                              void* d_out, int out_size, void* d_ws, size_t ws_size,
                              hipStream_t stream) {
    (void)in_sizes; (void)n_in; (void)out_size; (void)ws_size;
    const float* p      = (const float*)d_in[0];
    const unsigned char* p_mask = (const unsigned char*)d_in[1];
    const float* q      = (const float*)d_in[2];
    const unsigned char* q_mask = (const unsigned char*)d_in[3];
    const float* rep    = (const float*)d_in[4];
    const float* rep_p  = (const float*)d_in[5];
    const float* U_w    = (const float*)d_in[7];
    const float* D      = (const float*)d_in[8];
    float* out          = (float*)d_out;

    // Workspace layout (bytes)
    char* ws = (char*)d_ws;
    __bf16* Up    = (__bf16*)(ws);                               // 32 MiB
    __bf16* Uq    = (__bf16*)(ws + (size_t)33554432);            // 32 MiB
    __bf16* S     = (__bf16*)(ws + (size_t)67108864);            // 64 MiB
    float* stats1 = (float*)(ws + (size_t)134217728);            // 128 KiB
    float* stats2 = (float*)(ws + (size_t)134217728 + 131072);   // 128 KiB
    float* part2  = (float*)(ws + (size_t)134217728 + 262144);   // 1 MiB

    dim3 blk(256);
    proj_kernel<<<dim3(BB * (LPN + LQN) / TM, HHN / TN), blk, 0, stream>>>(
        p, q, U_w, D, Up, Uq);
    scores_kernel<<<dim3(LPN / TM, LQN / TN, BB), blk, 0, stream>>>(Up, Uq, S);
    stats_row_kernel<<<dim3(BB * LPN / 8), blk, 0, stream>>>(S, q_mask, stats1);
    stats_col_partial_kernel<<<dim3(LQN / 256, LPN / 256, BB), blk, 0, stream>>>(
        S, p_mask, part2);
    stats_col_combine_kernel<<<dim3(BB * LQN / 256), blk, 0, stream>>>(part2, stats2);
    // p -> q direction: output[b, Lp, Drep]
    attn_kernel<<<dim3(LPN / TM, DREPN / TN, BB), blk, 0, stream>>>(
        S, stats1, q_mask, rep, out, 0);
    // q -> p direction: output_q[b, Lq, Drep] (second half of d_out)
    attn_kernel<<<dim3(LQN / TM, DREPN / TN, BB), blk, 0, stream>>>(
        S, stats2, p_mask, rep_p, out + (size_t)BB * LPN * DREPN, 1);
}
